// QKVAttentionLegacy_2559800508521
// MI455X (gfx1250) — matrix operand order repaired and checked by reading, not compiled, not run on hardware
//
#include <hip/hip_runtime.h>

typedef __attribute__((ext_vector_type(16))) _Float16 v16h;
typedef __attribute__((ext_vector_type(8)))  _Float16 v8h;
typedef __attribute__((ext_vector_type(8)))  float    v8f;

namespace {
constexpr int kBS    = 16;
constexpr int kHeads = 8;
constexpr int kB     = kBS * kHeads;   // 128 flat heads
constexpr int kCh    = 64;             // head dim
constexpr int kT     = 1024;           // query length
constexpr int kSEnc  = 256;            // encoder kv length
constexpr int kSTot  = 1280;           // total kv length
constexpr int kTileT = 128;            // query rows per workgroup

constexpr size_t kQBytes = (size_t)kB * kCh * kT * 2;     // 16.78 MB (A-swizzled f16 Q)
constexpr size_t kKBytes = (size_t)kB * kSTot * kCh * 2;  // 20.97 MB ([s][c] f16 K)
constexpr size_t kVBytes = (size_t)kB * kCh * kSTot * 2;  // 20.97 MB ([c][s] f16 V)
}

__device__ inline v16h hcat(v8h a, v8h b) {
    return __builtin_shufflevector(a, b, 0,1,2,3,4,5,6,7,8,9,10,11,12,13,14,15);
}

// Issue one per-lane 16B async copy global -> LDS (ASYNCcnt-tracked DMA path).
__device__ inline void async_cp16(unsigned lds_off, const _Float16* g) {
    unsigned long long ga = (unsigned long long)(size_t)g;
    asm volatile("global_load_async_to_lds_b128 %0, %1, off"
                 :: "v"(lds_off), "v"(ga) : "memory");
}
__device__ inline void wait_async0() {
    asm volatile("s_wait_asynccnt 0x0" ::: "memory");
}
#define LDSOFF(p) ((unsigned)(size_t)(p))

// ============================== pass 1: f32 -> fused/scaled f16 pack =========

// Q packed directly in WMMA A-operand order:
// qp[b][tile][part][lane][j], tile = t/16, part = c-half, c = 32*part + 16*(lane/16) + j.
__global__ __launch_bounds__(256)
void pack_q(const float* __restrict__ qkv, const float* __restrict__ pe,
            _Float16* __restrict__ qp)
{
    const int b = blockIdx.y, batch = b >> 3, head = b & 7;
    const float* qg = qkv + (size_t)batch * 1536 * kT + (size_t)head * 192 * kT;
    const int tile = blockIdx.x;                    // 64 tiles of 16 query rows
    const int o0 = threadIdx.x * 4;
    _Float16* dst = qp + ((size_t)b * 64 + tile) * 1024 + o0;
    #pragma unroll
    for (int e = 0; e < 4; ++e) {
        const int o = o0 + e;
        const int part = o >> 9, lane = (o >> 4) & 31, j = o & 15;
        const int lh = lane & 15, h = lane >> 4;
        const int t  = tile * 16 + lh;
        const int cj = 32 * part + 16 * h + j;
        dst[e] = (_Float16)((qg[(size_t)cj * kT + t] + pe[(size_t)cj * kT + t]) * 0.125f);
    }
}

// K packed [s][c] (c contiguous), pe fused, encoder part concatenated in front.
__global__ __launch_bounds__(256)
void pack_k(const float* __restrict__ qkv, const float* __restrict__ ekv,
            const float* __restrict__ pe, _Float16* __restrict__ kp)
{
    const int b = blockIdx.y, batch = b >> 3, head = b & 7;
    const float* kg  = qkv + (size_t)batch * 1536 * kT + (size_t)head * 192 * kT + (size_t)kCh * kT;
    const float* ekg = ekv + (size_t)batch * 1024 * kSEnc + (size_t)head * 128 * kSEnc;
    const size_t f0 = ((size_t)blockIdx.x * 256 + threadIdx.x) * 4;
    _Float16* dst = kp + (size_t)b * kSTot * kCh + f0;
    #pragma unroll
    for (int e = 0; e < 4; ++e) {
        const size_t f = f0 + e;
        const int c = (int)(f & 63), s = (int)(f >> 6);
        float v;
        if (s < kSEnc) v = ekg[(size_t)c * kSEnc + s];
        else           v = kg[(size_t)c * kT + (s - kSEnc)] + pe[(size_t)c * kT + (s - kSEnc)];
        dst[e] = (_Float16)v;
    }
}

// V packed [c][s] (s contiguous), encoder part concatenated in front.
__global__ __launch_bounds__(256)
void pack_v(const float* __restrict__ qkv, const float* __restrict__ ekv,
            _Float16* __restrict__ vp)
{
    const int b = blockIdx.y, batch = b >> 3, head = b & 7;
    const float* vg  = qkv + (size_t)batch * 1536 * kT + (size_t)head * 192 * kT + (size_t)(2 * kCh) * kT;
    const float* evg = ekv + (size_t)batch * 1024 * kSEnc + (size_t)head * 128 * kSEnc + (size_t)kCh * kSEnc;
    const size_t f0 = ((size_t)blockIdx.x * 256 + threadIdx.x) * 4;
    _Float16* dst = vp + (size_t)b * kCh * kSTot + f0;
    #pragma unroll
    for (int e = 0; e < 4; ++e) {
        const size_t f = f0 + e;
        const int c = (int)(f / kSTot), s = (int)(f % kSTot);
        dst[e] = (_Float16)(s < kSEnc ? evg[(size_t)c * kSEnc + s]
                                      : vg[(size_t)c * kT + (s - kSEnc)]);
    }
}

// ============================== pass 2: flash attention, async double-buffer ==

__global__ __launch_bounds__(256)
void attn2(const _Float16* __restrict__ qp, const _Float16* __restrict__ kp,
           const _Float16* __restrict__ vp, float* __restrict__ out)
{
    // 144B rows (72 halves): 16B-aligned segments, conflict-free b128 DS reads.
    __shared__ __align__(16) _Float16 kbuf[2][64][72];   // [s][c]
    __shared__ __align__(16) _Float16 vbuf[2][64][72];   // [c][s]
    __shared__ __align__(16) _Float16 pls[8][16][72];    // per-wave P relayout

    const int tid  = threadIdx.x;
    const int lane = tid & 31;
    const int wv   = tid >> 5;
    const int lh   = lane & 15;
    const int h    = lane >> 4;

    const int bh    = blockIdx.y;
    const int batch = bh >> 3;
    const int head  = bh & 7;
    const int t0    = blockIdx.x * kTileT;
    const int tile  = blockIdx.x * 8 + wv;

    const _Float16* kph = kp + (size_t)bh * kSTot * kCh;
    const _Float16* vph = vp + (size_t)bh * kCh * kSTot;

    // Q tile: two 32B vector loads straight into A-operand registers.
    const _Float16* qt = qp + ((size_t)bh * 64 + tile) * 1024;
    const v16h qa0 = *(const v16h*)(qt + lane * 16);
    const v16h qa1 = *(const v16h*)(qt + 512 + lane * 16);

    float m_i[8], l_i[8];
    v8f o_acc[4];
    #pragma unroll
    for (int r = 0; r < 8; ++r) { m_i[r] = -1e30f; l_i[r] = 0.0f; }
    #pragma unroll
    for (int cc = 0; cc < 4; ++cc) o_acc[cc] = (v8f){0,0,0,0,0,0,0,0};

    // Each wave issues 2 K + 2 V async instructions per chunk (64x128B each tile).
    auto stage = [&](int kc, int bi) {
        const int s0 = kc * 64;
        #pragma unroll
        for (int u = 0; u < 2; ++u) {
            const int idx = (wv * 2 + u) * 32 + lane;    // 0..511
            const int r = idx >> 3, seg = idx & 7;
            async_cp16(LDSOFF(&kbuf[bi][r][seg * 8]),
                       kph + (size_t)(s0 + r) * kCh + seg * 8);
            async_cp16(LDSOFF(&vbuf[bi][r][seg * 8]),
                       vph + (size_t)r * kSTot + s0 + seg * 8);
        }
    };

    stage(0, 0);
    wait_async0();
    __syncthreads();

    for (int kc = 0; kc < kSTot / 64; ++kc) {
        const int bi = kc & 1;
        if (kc + 1 < kSTot / 64) stage(kc + 1, bi ^ 1);   // overlap DMA with compute

        // ---- logits S(16x64): 4 s-tiles, k-dim chained 2x32
        v8f acc[4];
        #pragma unroll
        for (int st = 0; st < 4; ++st) {
            const v8h* kr = (const v8h*)&kbuf[bi][st * 16 + lh][0];
            const v16h kb0 = hcat(kr[2 * h],     kr[2 * h + 1]);
            const v16h kb1 = hcat(kr[4 + 2 * h], kr[5 + 2 * h]);
            v8f a = (v8f){0,0,0,0,0,0,0,0};
            a = __builtin_amdgcn_wmma_f32_16x16x32_f16(false, qa0, false, kb0, (short)0, a, false, false);
            a = __builtin_amdgcn_wmma_f32_16x16x32_f16(false, qa1, false, kb1, (short)0, a, false, false);
            acc[st] = a;
        }

        // ---- online softmax (rows live across the 16 lanes of each half)
        float alpha[8];
        #pragma unroll
        for (int r = 0; r < 8; ++r) {
            float mx = fmaxf(fmaxf(acc[0][r], acc[1][r]), fmaxf(acc[2][r], acc[3][r]));
            #pragma unroll
            for (int m = 8; m >= 1; m >>= 1)
                mx = fmaxf(mx, __shfl_xor(mx, m, 32));
            const float mn = fmaxf(m_i[r], mx);
            alpha[r] = __expf(m_i[r] - mn);
            float rs = 0.0f;
            #pragma unroll
            for (int st = 0; st < 4; ++st) {
                acc[st][r] = __expf(acc[st][r] - mn);
                rs += acc[st][r];
            }
            #pragma unroll
            for (int m = 8; m >= 1; m >>= 1)
                rs += __shfl_xor(rs, m, 32);
            l_i[r] = l_i[r] * alpha[r] + rs;
            m_i[r] = mn;
        }

        // ---- P: C-layout f32 -> A-layout f16 via per-wave LDS (DS in-order per wave)
        #pragma unroll
        for (int r = 0; r < 8; ++r)
            #pragma unroll
            for (int st = 0; st < 4; ++st)
                pls[wv][r + 8 * h][st * 16 + lh] = (_Float16)acc[st][r];
        const v8h* pr = (const v8h*)&pls[wv][lh][0];
        const v16h pa0 = hcat(pr[2 * h],     pr[2 * h + 1]);
        const v16h pa1 = hcat(pr[4 + 2 * h], pr[5 + 2 * h]);

        // ---- O(16x64) += P(16x64) * V^T(64x64): 4 channel tiles, k chained 2x32
        #pragma unroll
        for (int cc = 0; cc < 4; ++cc) {
            v8f o = o_acc[cc];
            #pragma unroll
            for (int r = 0; r < 8; ++r) o[r] *= alpha[r];
            const v8h* vr = (const v8h*)&vbuf[bi][cc * 16 + lh][0];
            const v16h vb0 = hcat(vr[2 * h],     vr[2 * h + 1]);
            const v16h vb1 = hcat(vr[4 + 2 * h], vr[5 + 2 * h]);
            o = __builtin_amdgcn_wmma_f32_16x16x32_f16(false, pa0, false, vb0, (short)0, o, false, false);
            o = __builtin_amdgcn_wmma_f32_16x16x32_f16(false, pa1, false, vb1, (short)0, o, false, false);
            o_acc[cc] = o;
        }

        wait_async0();       // next buffer staged
        __syncthreads();     // everyone done reading current buffer
    }

    // ---- epilogue
    float inv[8];
    #pragma unroll
    for (int r = 0; r < 8; ++r) inv[r] = 1.0f / l_i[r];
    float* ob = out + (size_t)batch * 512 * kT + (size_t)head * kCh * kT;
    #pragma unroll
    for (int cc = 0; cc < 4; ++cc) {
        const int c = cc * 16 + lh;
        #pragma unroll
        for (int r = 0; r < 8; ++r) {
            const int t = t0 + wv * 16 + r + 8 * h;
            ob[(size_t)c * kT + t] = o_acc[cc][r] * inv[r];
        }
    }
}

// ============================== fallback: fused kernel (round-1, no workspace) =

__global__ __launch_bounds__(256)
void attn_fused(const float* __restrict__ qkv, const float* __restrict__ ekv,
                const float* __restrict__ pe, float* __restrict__ out)
{
    __shared__ _Float16 kls[32][72];
    __shared__ _Float16 vls[kCh][40];
    __shared__ _Float16 pls[8][16][32];

    const int tid = threadIdx.x, lane = tid & 31, wv = tid >> 5;
    const int lh = lane & 15, h = lane >> 4;
    const int bh = blockIdx.y, batch = bh >> 3, head = bh & 7;
    const int t0 = blockIdx.x * kTileT;

    const float* qkv_b = qkv + (size_t)batch * 1536 * kT + (size_t)head * 192 * kT;
    const float* qg = qkv_b;
    const float* kg = qkv_b + (size_t)kCh * kT;
    const float* vg = qkv_b + (size_t)2 * kCh * kT;
    const float* ekg = ekv + (size_t)batch * 1024 * kSEnc + (size_t)head * 128 * kSEnc;
    const float* evg = ekg + (size_t)kCh * kSEnc;

    v16h qa0, qa1;
    {
        const int t = t0 + wv * 16 + lh;
        #pragma unroll
        for (int j = 0; j < 16; ++j) {
            const int cj = 16 * h + j;
            const float q0 = qg[(size_t)cj * kT + t]        + pe[(size_t)cj * kT + t];
            const float q1 = qg[(size_t)(cj + 32) * kT + t] + pe[(size_t)(cj + 32) * kT + t];
            qa0[j] = (_Float16)(q0 * 0.125f);
            qa1[j] = (_Float16)(q1 * 0.125f);
        }
    }

    float m_i[8], l_i[8];
    v8f o_acc[4];
    #pragma unroll
    for (int r = 0; r < 8; ++r) { m_i[r] = -1e30f; l_i[r] = 0.0f; }
    #pragma unroll
    for (int cc = 0; cc < 4; ++cc) o_acc[cc] = (v8f){0,0,0,0,0,0,0,0};

    for (int kc = 0; kc < kSTot / 32; ++kc) {
        const int s0 = kc * 32;
        {
            const int so = tid & 31, s = s0 + so;
            #pragma unroll
            for (int p = 0; p < 8; ++p) {
                const int c = (tid >> 5) + p * 8;
                float kval, vval;
                if (s < kSEnc) { kval = ekg[(size_t)c * kSEnc + s]; vval = evg[(size_t)c * kSEnc + s]; }
                else { const int sq = s - kSEnc;
                       kval = kg[(size_t)c * kT + sq] + pe[(size_t)c * kT + sq];
                       vval = vg[(size_t)c * kT + sq]; }
                kls[so][c] = (_Float16)kval;
                vls[c][so] = (_Float16)vval;
            }
        }
        __syncthreads();

        v8f acc0 = (v8f){0,0,0,0,0,0,0,0}, acc1 = (v8f){0,0,0,0,0,0,0,0};
        {
            v16h kb0, kb1;
            #pragma unroll
            for (int j = 0; j < 16; ++j) { kb0[j] = kls[lh][16 * h + j]; kb1[j] = kls[lh][32 + 16 * h + j]; }
            acc0 = __builtin_amdgcn_wmma_f32_16x16x32_f16(false, qa0, false, kb0, (short)0, acc0, false, false);
            acc0 = __builtin_amdgcn_wmma_f32_16x16x32_f16(false, qa1, false, kb1, (short)0, acc0, false, false);
            #pragma unroll
            for (int j = 0; j < 16; ++j) { kb0[j] = kls[16 + lh][16 * h + j]; kb1[j] = kls[16 + lh][32 + 16 * h + j]; }
            acc1 = __builtin_amdgcn_wmma_f32_16x16x32_f16(false, qa0, false, kb0, (short)0, acc1, false, false);
            acc1 = __builtin_amdgcn_wmma_f32_16x16x32_f16(false, qa1, false, kb1, (short)0, acc1, false, false);
        }

        float alpha[8];
        #pragma unroll
        for (int r = 0; r < 8; ++r) {
            float mx = fmaxf(acc0[r], acc1[r]);
            #pragma unroll
            for (int m = 8; m >= 1; m >>= 1) mx = fmaxf(mx, __shfl_xor(mx, m, 32));
            const float mn = fmaxf(m_i[r], mx);
            alpha[r] = __expf(m_i[r] - mn);
            acc0[r] = __expf(acc0[r] - mn);
            acc1[r] = __expf(acc1[r] - mn);
            float rs = acc0[r] + acc1[r];
            #pragma unroll
            for (int m = 8; m >= 1; m >>= 1) rs += __shfl_xor(rs, m, 32);
            l_i[r] = l_i[r] * alpha[r] + rs;
            m_i[r] = mn;
        }

        #pragma unroll
        for (int r = 0; r < 8; ++r) {
            pls[wv][r + 8 * h][lh]      = (_Float16)acc0[r];
            pls[wv][r + 8 * h][16 + lh] = (_Float16)acc1[r];
        }
        v16h pa;
        #pragma unroll
        for (int j = 0; j < 16; ++j) {
            const int so = 16 * h + j;
            pa[j] = pls[wv][lh][so];
        }

        #pragma unroll
        for (int cc = 0; cc < 4; ++cc) {
            v8f o = o_acc[cc];
            #pragma unroll
            for (int r = 0; r < 8; ++r) o[r] *= alpha[r];
            v16h vb;
            #pragma unroll
            for (int j = 0; j < 16; ++j) vb[j] = vls[cc * 16 + lh][16 * h + j];
            o_acc[cc] = __builtin_amdgcn_wmma_f32_16x16x32_f16(false, pa, false, vb, (short)0, o, false, false);
        }
        __syncthreads();
    }

    float inv[8];
    #pragma unroll
    for (int r = 0; r < 8; ++r) inv[r] = 1.0f / l_i[r];
    float* ob = out + (size_t)batch * 512 * kT + (size_t)head * kCh * kT;
    #pragma unroll
    for (int cc = 0; cc < 4; ++cc) {
        const int c = cc * 16 + lh;
        #pragma unroll
        for (int r = 0; r < 8; ++r) {
            const int t = t0 + wv * 16 + r + 8 * h;
            ob[(size_t)c * kT + t] = o_acc[cc][r] * inv[r];
        }
    }
}

// ============================== host ==========================================

extern "C" void kernel_launch(void* const* d_in, const int* in_sizes, int n_in,
                              void* d_out, int out_size, void* d_ws, size_t ws_size,
                              hipStream_t stream) {
    (void)in_sizes; (void)n_in; (void)out_size;
    const float* qkv = (const float*)d_in[0];   // (16, 1536, 1024) f32
    const float* ekv = (const float*)d_in[1];   // (16, 1024, 256)  f32
    const float* pe  = (const float*)d_in[2];   // (64, 1024)       f32
    float* out = (float*)d_out;                 // (16, 512, 1024)  f32

    if (ws_size >= kQBytes + kKBytes + kVBytes && d_ws != nullptr) {
        char* w = (char*)d_ws;
        _Float16* qp = (_Float16*)w;
        _Float16* kp = (_Float16*)(w + kQBytes);
        _Float16* vp = (_Float16*)(w + kQBytes + kKBytes);
        pack_q<<<dim3(64, kB), 256, 0, stream>>>(qkv, pe, qp);
        pack_k<<<dim3(80, kB), 256, 0, stream>>>(qkv, ekv, pe, kp);
        pack_v<<<dim3(80, kB), 256, 0, stream>>>(qkv, ekv, vp);
        attn2<<<dim3(kT / kTileT, kB), 256, 0, stream>>>(qp, kp, vp, out);
    } else {
        attn_fused<<<dim3(kT / kTileT, kB), 256, 0, stream>>>(qkv, ekv, pe, out);
    }
}
